// Decoder_84009560309891
// MI455X (gfx1250) — matrix operand order repaired and checked by reading, not compiled, not run on hardware
//
#include <hip/hip_runtime.h>
#include <hip/hip_bf16.h>
#include <math.h>

typedef __attribute__((ext_vector_type(16))) _Float16 v16h;
typedef __attribute__((ext_vector_type(8)))  float    v8f;

#define BB      16
#define LATENT  64
#define NSTATS  7
#define HDIM    128
#define NNODE   256
#define NPAIR   32640u                 // N*(N-1)/2
#define NRAND   (BB * NPAIR * 2u)      // 1,044,480 threefry words
#define NHALF   (NRAND / 2u)          // 522,240

// ---------------------------------------------------------------------------
// Kernel A: single-wave WMMA kernel computing per-batch dlogit[16].
// The latent is broadcast to all 256 nodes, so the entire node MLP collapses
// to 16-row GEMMs executed with v_wmma_f32_16x16x32_f16 (M=16 == batch).
// ---------------------------------------------------------------------------

// GEMM: Out[16 x N] = A[16 x KREAL] * W[KREAL x N].
//  A: LDS, row-major, leading dim = lda (zero-padded to KPAD).
//  W: global, row-major [KREAL x N].
// Operand layouts per ISA 7.12.2 (wave32):
//  A (16-bit, 16x32): lanes 0-15 -> M=lane, K-halfbase 0; lanes 16-31 -> +8.
//     element j: K = 16*(j>>3) + halfbase + (j&7)
//  B (16-bit, 32x16): same K order as A: lanes 0-15 K-halfbase 0; lanes 16-31 -> +8;
//     element j: K = 16*(j>>3) + halfbase + (j&7) ; N = lane&15
//  D (f32, 16x16): VGPR v -> M = v + 8*(lane>>4), N = lane&15
template <int KREAL, int KPAD, int N>
__device__ inline void gemm16(const float* A, int lda,
                              const float* __restrict__ W,
                              float* Out, int ldo) {
    const int lane  = threadIdx.x;
    const int nloc  = lane & 15;
    const int half8 = (lane >> 4) << 3;   // A K-halfbase and D M-base
    constexpr int NK = KPAD / 32;

    // Preload all A fragments once (invariant across N-tiles).
    v16h afr[NK];
#pragma unroll
    for (int kc = 0; kc < NK; ++kc) {
#pragma unroll
        for (int j = 0; j < 16; ++j) {
            int k = kc * 32 + ((j >> 3) << 4) + half8 + (j & 7);
            afr[kc][j] = (_Float16)A[nloc * lda + k];
        }
    }

    const int kb0 = (lane >> 4) << 4;     // unused; B takes the same k order as A
#pragma unroll
    for (int n0 = 0; n0 < N; n0 += 16) {
        v8f acc = {0.f, 0.f, 0.f, 0.f, 0.f, 0.f, 0.f, 0.f};
#pragma unroll
        for (int kc = 0; kc < NK; ++kc) {
            v16h b;
#pragma unroll
            for (int j = 0; j < 16; ++j) {
                int k = kc * 32 + ((j >> 3) << 4) + half8 + (j & 7);
                if constexpr (KREAL == KPAD) {
                    b[j] = (_Float16)W[k * N + n0 + nloc];
                } else {
                    // Branchless tail guard: clamped load + VALU select.
                    int kk   = (k < KREAL) ? k : 0;
                    float wv = W[kk * N + n0 + nloc];
                    b[j] = (_Float16)((k < KREAL) ? wv : 0.0f);
                }
            }
            acc = __builtin_amdgcn_wmma_f32_16x16x32_f16(
                false, afr[kc], false, b, (short)0, acc, false, false);
        }
#pragma unroll
        for (int v = 0; v < 8; ++v)
            Out[(half8 + v) * ldo + n0 + nloc] = acc[v];
    }
}

// bias (optional) + relu (optional) + layernorm (optional) over buf[16 x C]
__device__ inline void postproc(float* buf, int C,
                                const float* __restrict__ bias, bool do_relu,
                                const float* __restrict__ g,
                                const float* __restrict__ be,
                                float* stat) {
    const int lane = threadIdx.x;
    if (bias != nullptr || do_relu) {
        for (int t = lane; t < 16 * C; t += 32) {
            float v = buf[t];
            if (bias) v += bias[t % C];
            if (do_relu) v = fmaxf(v, 0.0f);
            buf[t] = v;
        }
        __syncthreads();
    }
    if (g != nullptr) {
        if (lane < 16) {
            float s = 0.f, s2 = 0.f;
            for (int c = 0; c < C; ++c) {
                float v = buf[lane * C + c];
                s += v; s2 += v * v;
            }
            float m   = s / (float)C;
            float var = fmaxf(s2 / (float)C - m * m, 0.0f);
            stat[lane]      = m;
            stat[16 + lane] = 1.0f / sqrtf(var + 1e-5f);
        }
        __syncthreads();
        for (int t = lane; t < 16 * C; t += 32) {
            int r = t / C, c = t % C;
            buf[t] = (buf[t] - stat[r]) * stat[16 + r] * g[c] + be[c];
        }
        __syncthreads();
    }
}

__global__ __launch_bounds__(32)
void decoder_node_edge(const float* __restrict__ x,
                       const float* __restrict__ stats,
                       const float* __restrict__ ip_w,  const float* __restrict__ ip_b,
                       const float* __restrict__ ip_g,  const float* __restrict__ ip_be,
                       const float* __restrict__ dr_w,  const float* __restrict__ dr_b,
                       const float* __restrict__ dr_g,  const float* __restrict__ dr_be,
                       const float* __restrict__ r1w,   const float* __restrict__ r1b,
                       const float* __restrict__ r2w,   const float* __restrict__ r2b,
                       const float* __restrict__ n1g,   const float* __restrict__ n1b,
                       const float* __restrict__ n2g,   const float* __restrict__ n2b,
                       const float* __restrict__ e1w,   const float* __restrict__ e1b,
                       const float* __restrict__ elg,   const float* __restrict__ elb,
                       const float* __restrict__ e2w,   const float* __restrict__ e2b,
                       float* __restrict__ dlogit) {
    __shared__ float zbuf[16 * 96];        // z, padded K 71 -> 96
    __shared__ float bufA[16 * 256];       // h1 / duplicated edge input
    __shared__ float bufB[16 * 128];       // h
    __shared__ float bufC[16 * 128];       // scratch r
    __shared__ float bufD[16 * 128];       // scratch r2
    __shared__ float stat[32];

    const int lane = threadIdx.x;

    // pull weight streams toward GL2 early (global_prefetch_b8)
    if (lane == 0) {
        __builtin_prefetch(ip_w, 0, 1);
        __builtin_prefetch(dr_w, 0, 1);
        __builtin_prefetch(r1w, 0, 1);
        __builtin_prefetch(e1w, 0, 1);
    }

    // build z = concat(x, stats), zero-pad columns 71..95
    for (int t = lane; t < 16 * 96; t += 32) {
        int r = t / 96, c = t % 96;
        float v = 0.0f;
        if (c < LATENT)                 v = x[r * LATENT + c];
        else if (c < LATENT + NSTATS)   v = stats[r * NSTATS + (c - LATENT)];
        zbuf[t] = v;
    }
    __syncthreads();

    // input_proj: [16x71] @ [71x256] -> relu -> LN
    gemm16<LATENT + NSTATS, 96, 256>(zbuf, 96, ip_w, bufA, 256);
    __syncthreads();
    postproc(bufA, 256, ip_b, true, ip_g, ip_be, stat);

    // dim_reduce: [16x256] @ [256x128] -> relu -> LN
    gemm16<256, 256, 128>(bufA, 256, dr_w, bufB, 128);
    __syncthreads();
    postproc(bufB, 128, dr_b, true, dr_g, dr_be, stat);

    // residual blocks: h = 1.1*h + 0.1*(r + bias)
    for (int l = 0; l < 2; ++l) {
        const float* W1 = r1w + l * HDIM * HDIM;
        const float* W2 = r2w + l * HDIM * HDIM;

        for (int t = lane; t < 16 * HDIM; t += 32) bufC[t] = bufB[t];
        __syncthreads();
        postproc(bufC, HDIM, nullptr, false, n1g + l * HDIM, n1b + l * HDIM, stat);

        gemm16<HDIM, HDIM, HDIM>(bufC, HDIM, W1, bufD, HDIM);
        __syncthreads();
        postproc(bufD, HDIM, r1b + l * HDIM, true, n2g + l * HDIM, n2b + l * HDIM, stat);

        gemm16<HDIM, HDIM, HDIM>(bufD, HDIM, W2, bufC, HDIM);
        __syncthreads();
        for (int t = lane; t < 16 * HDIM; t += 32) {
            float r = bufC[t] + r2b[l * HDIM + (t % HDIM)];
            bufB[t] = 1.1f * bufB[t] + 0.1f * r;
        }
        __syncthreads();
    }

    // attention over identical nodes == uniform 1/N; edge input = [h, h]
    for (int t = lane; t < 16 * 256; t += 32) {
        int r = t / 256, c = t % 256;
        bufA[t] = bufB[r * HDIM + (c & 127)] * (1.0f / (float)NNODE);
    }
    __syncthreads();

    // edge_mlp layer 1: [16x256] @ [256x128] -> relu -> LN
    gemm16<256, 256, 128>(bufA, 256, e1w, bufC, 128);
    __syncthreads();
    postproc(bufC, 128, e1b, true, elg, elb, stat);

    // logits head (128 -> 2): only the decision margin logit0 - logit1 matters
    if (lane < 16) {
        float s = e2b[0] - e2b[1];
        for (int k = 0; k < HDIM; ++k)
            s += bufC[lane * HDIM + k] * (e2w[2 * k] - e2w[2 * k + 1]);
        dlogit[lane] = s;
    }
}

// ---------------------------------------------------------------------------
// Kernel B: output-bound adjacency kernel. One thread per adj element.
// Exact JAX threefry2x32 (key = (0,42), 20 rounds) -> uniform -> gumbel.
// ---------------------------------------------------------------------------
__device__ inline unsigned rotl32(unsigned v, int r) {
    return (v << r) | (v >> (32 - r));
}

__device__ inline unsigned threefry_word(unsigned idx) {
    // JAX splits the flat counter in halves: out[i] = enc(i, i+half).x0 for
    // i < half, out[half+i] = enc(i, i+half).x1.
    const bool lo = idx < NHALF;
    unsigned x0 = lo ? idx : idx - NHALF;
    unsigned x1 = x0 + NHALF;
    const unsigned k0 = 0u, k1 = 42u;
    const unsigned k2 = 0x1BD11BDAu ^ k0 ^ k1;
    x0 += k0; x1 += k1;
#pragma unroll
    for (int r = 0; r < 4; ++r) { const int ra[4] = {13,15,26,6};
        x0 += x1; x1 = rotl32(x1, ra[r]); x1 ^= x0; }
    x0 += k1; x1 += k2 + 1u;
#pragma unroll
    for (int r = 0; r < 4; ++r) { const int rb[4] = {17,29,16,24};
        x0 += x1; x1 = rotl32(x1, rb[r]); x1 ^= x0; }
    x0 += k2; x1 += k0 + 2u;
#pragma unroll
    for (int r = 0; r < 4; ++r) { const int ra[4] = {13,15,26,6};
        x0 += x1; x1 = rotl32(x1, ra[r]); x1 ^= x0; }
    x0 += k0; x1 += k1 + 3u;
#pragma unroll
    for (int r = 0; r < 4; ++r) { const int rb[4] = {17,29,16,24};
        x0 += x1; x1 = rotl32(x1, rb[r]); x1 ^= x0; }
    x0 += k1; x1 += k2 + 4u;
#pragma unroll
    for (int r = 0; r < 4; ++r) { const int ra[4] = {13,15,26,6};
        x0 += x1; x1 = rotl32(x1, ra[r]); x1 ^= x0; }
    x0 += k2; x1 += k0 + 5u;
    return lo ? x0 : x1;
}

__device__ inline float gumbel_at(unsigned idx) {
    unsigned bits = threefry_word(idx);
    float f = __uint_as_float((bits >> 9) | 0x3f800000u) - 1.0f;   // [0,1)
    float u = fmaxf(f * (1.0f - 1e-9f) + 1e-9f, 1e-9f);            // jax uniform
    return -logf(-logf(u));
}

__global__ __launch_bounds__(256)
void decoder_adj(const float* __restrict__ dlogit, float* __restrict__ out) {
    unsigned tid = blockIdx.x * 256u + threadIdx.x;   // b*65536 + i*256 + j
    unsigned b = tid >> 16;
    unsigned i = (tid >> 8) & 255u;
    unsigned j = tid & 255u;
    float val = 0.0f;
    if (i != j) {
        unsigned ii = (i < j) ? i : j;
        unsigned jj = (i < j) ? j : i;
        // triu pair index: offset(ii) + (jj - ii - 1), offset = ii*(2N-1-ii)/2
        unsigned p    = (ii * (2u * NNODE - 1u - ii)) / 2u + (jj - ii - 1u);
        unsigned base = (b * NPAIR + p) * 2u;
        float g0 = gumbel_at(base);
        float g1 = gumbel_at(base + 1u);
        // hard straight-through: argmax picks channel 0 on ties
        val = (dlogit[b] + g0 - g1 >= 0.0f) ? 1.0f : 0.0f;
    }
    out[tid] = val;   // fully coalesced 4 MB stream
}

// ---------------------------------------------------------------------------
extern "C" void kernel_launch(void* const* d_in, const int* in_sizes, int n_in,
                              void* d_out, int out_size, void* d_ws, size_t ws_size,
                              hipStream_t stream) {
    const float* x      = (const float*)d_in[0];
    const float* stats  = (const float*)d_in[1];
    const float* ip_w   = (const float*)d_in[2];
    const float* ip_b   = (const float*)d_in[3];
    const float* ip_g   = (const float*)d_in[4];
    const float* ip_be  = (const float*)d_in[5];
    const float* dr_w   = (const float*)d_in[6];
    const float* dr_b   = (const float*)d_in[7];
    const float* dr_g   = (const float*)d_in[8];
    const float* dr_be  = (const float*)d_in[9];
    const float* r1w    = (const float*)d_in[10];
    const float* r1b    = (const float*)d_in[11];
    const float* r2w    = (const float*)d_in[12];
    const float* r2b    = (const float*)d_in[13];
    const float* n1g    = (const float*)d_in[14];
    const float* n1b    = (const float*)d_in[15];
    const float* n2g    = (const float*)d_in[16];
    const float* n2b    = (const float*)d_in[17];
    // d_in[18], d_in[19]: attn_w / attn_b — provably unused (uniform softmax)
    const float* e1w    = (const float*)d_in[20];
    const float* e1b    = (const float*)d_in[21];
    const float* elg    = (const float*)d_in[22];
    const float* elb    = (const float*)d_in[23];
    const float* e2w    = (const float*)d_in[24];
    const float* e2b    = (const float*)d_in[25];
    // d_in[26]: temperature — positive scalar, argmax-invariant

    float* dlogit = (float*)d_ws;     // 16 floats of scratch

    decoder_node_edge<<<1, 32, 0, stream>>>(
        x, stats, ip_w, ip_b, ip_g, ip_be, dr_w, dr_b, dr_g, dr_be,
        r1w, r1b, r2w, r2b, n1g, n1b, n2g, n2b,
        e1w, e1b, elg, elb, e2w, e2b, dlogit);

    const int total = BB * NNODE * NNODE;             // 1,048,576 elements
    decoder_adj<<<total / 256, 256, 0, stream>>>(dlogit, (float*)d_out);
}
